// FactorizedRandomAttention_3418793968316
// MI455X (gfx1250) — matrix operand order repaired and checked by reading, not compiled, not run on hardware
//
#include <hip/hip_runtime.h>
#include <hip/hip_bf16.h>

typedef __attribute__((ext_vector_type(16))) _Float16 v16h;
typedef __attribute__((ext_vector_type(8)))  _Float16 v8h;
typedef __attribute__((ext_vector_type(2)))  _Float16 v2h;
typedef __attribute__((ext_vector_type(8)))  float    v8f;

// Problem constants (fixed by reference setup_inputs)
static constexpr int Bc  = 4;
static constexpr int Hc  = 8;
static constexpr int Lc  = 2048;
static constexpr int Rc  = 8;
static constexpr int DVc = 128;

static constexpr int ROWS   = 16;     // rows of attn per workgroup
static constexpr int NTHR   = 256;    // 8 waves (wave32)
static constexpr int KC     = 64;     // K-chunk per staging round (2 WMMA k-steps)
static constexpr int SS_STR = 2052;   // padded floats per score row (2052 % 64 == 4 -> no bank conflicts)
static constexpr int VT_STR = 72;     // padded halfs per v-column row (144B, 16B aligned; 36 dw stride)

// Dynamic LDS layout (bytes):
//   sS   : ROWS * SS_STR * 4   = 131328  (scores -> exp -> normalized attn, f32)
//   sVt  : DVc * VT_STR * 2    = 18432   (v chunk, f16, transposed [n][k], padded)
//   sA   : ROWS * Rc * 4       = 512     (a1 block)
//   sRed : ROWS * 16 * 4       = 1024    (softmax partials)
//   sRow : ROWS * 4            = 64      (row max / 1/sum)
static constexpr size_t OFF_VT  = (size_t)ROWS * SS_STR * 4;
static constexpr size_t OFF_A   = OFF_VT + (size_t)DVc * VT_STR * 2;
static constexpr size_t OFF_RED = OFF_A + (size_t)ROWS * Rc * 4;
static constexpr size_t OFF_ROW = OFF_RED + (size_t)ROWS * 16 * 4;
static constexpr size_t SMEM_BYTES = OFF_ROW + (size_t)ROWS * 4;

__global__ __launch_bounds__(NTHR, 1)
void fra_fused_kernel(const float* __restrict__ vmat,
                      const float* __restrict__ a1,
                      const float* __restrict__ a2,
                      const int*   __restrict__ mask,
                      float* __restrict__ out,
                      float* __restrict__ attn)
{
    extern __shared__ char smem[];
    float*    sS   = (float*)smem;                      // [ROWS][SS_STR]
    _Float16* sVt  = (_Float16*)(smem + OFF_VT);        // [DVc][VT_STR]
    float*    sA   = (float*)(smem + OFF_A);            // [ROWS][Rc]
    float*    sRed = (float*)(smem + OFF_RED);          // [ROWS][16]
    float*    sRow = (float*)(smem + OFF_ROW);          // [ROWS]

    const int tid  = threadIdx.x;
    const int wg   = blockIdx.x;
    const int ib   = wg % (Lc / ROWS);
    const int h    = (wg / (Lc / ROWS)) % Hc;
    const int b    = wg / ((Lc / ROWS) * Hc);
    const int i0   = ib * ROWS;
    const int bh   = b * Hc + h;

    const float* a2bh  = a2   + (size_t)bh * Rc * Lc;
    const int*   maskb = mask + (size_t)b * Lc * Lc;
    const float* vbh   = vmat + (size_t)bh * Lc * DVc;

    const int wave = tid >> 5;
    const int lane = tid & 31;

    // ---- Phase 0: load a1 block (16 rows x 8) ----
    if (tid < ROWS * Rc) {
        int r = tid / Rc, c = tid % Rc;
        sA[tid] = a1[((size_t)bh * Lc + i0 + r) * Rc + c];
    }
    __syncthreads();

    // ---- Phase 1: scores = a1_blk @ a2, masked, into LDS ----
    #pragma unroll
    for (int c = 0; c < Lc / NTHR; ++c) {
        const int j = tid + c * NTHR;
        float a2col[Rc];
        #pragma unroll
        for (int r = 0; r < Rc; ++r) a2col[r] = a2bh[(size_t)r * Lc + j];
        #pragma unroll
        for (int i = 0; i < ROWS; ++i) {
            float s = 0.0f;
            #pragma unroll
            for (int r = 0; r < Rc; ++r) s = fmaf(sA[i * Rc + r], a2col[r], s);
            const int mk = maskb[(size_t)(i0 + i) * Lc + j];
            sS[i * SS_STR + j] = mk ? s : -1000000000.0f;
        }
    }
    __syncthreads();

    // ---- Phase 2: row softmax over 2048 cols ----
    {
        const int i   = tid >> 4;    // row 0..15 (16 threads per row)
        const int l16 = tid & 15;

        float mx = -3.4e38f;
        #pragma unroll 8
        for (int c = 0; c < Lc / 16; ++c)
            mx = fmaxf(mx, sS[i * SS_STR + l16 + (c << 4)]);
        sRed[i * 16 + l16] = mx;
        __syncthreads();
        if (l16 == 0) {
            float m2 = sRed[i * 16];
            #pragma unroll
            for (int r = 1; r < 16; ++r) m2 = fmaxf(m2, sRed[i * 16 + r]);
            sRow[i] = m2;
        }
        __syncthreads();

        const float rmax = sRow[i];
        float sum = 0.0f;
        #pragma unroll 8
        for (int c = 0; c < Lc / 16; ++c) {
            const int idx = i * SS_STR + l16 + (c << 4);
            const float e = __expf(sS[idx] - rmax);
            sS[idx] = e;
            sum += e;
        }
        sRed[i * 16 + l16] = sum;
        __syncthreads();
        if (l16 == 0) {
            float s2 = sRed[i * 16];
            #pragma unroll
            for (int r = 1; r < 16; ++r) s2 += sRed[i * 16 + r];
            sRow[i] = 1.0f / s2;
        }
        __syncthreads();
    }

    // ---- Phase 2b: normalize in LDS + stream attn f32 to HBM (wave-contiguous 128B) ----
    {
        const int r0 = wave * 2;              // each wave owns 2 rows
        #pragma unroll
        for (int rr = 0; rr < 2; ++rr) {
            const int row = r0 + rr;
            const float inv = sRow[row];
            float* attnRow = attn + ((size_t)bh * Lc + i0 + row) * Lc;
            #pragma unroll 8
            for (int c = 0; c < Lc / 32; ++c) {
                const int j = lane + (c << 5);
                const float p = sS[row * SS_STR + j] * inv;
                sS[row * SS_STR + j] = p;     // keep normalized attn in LDS for WMMA
                attnRow[j] = p;               // 32 lanes x 4B = 128B contiguous
            }
        }
    }
    __syncthreads();

    // ---- Phase 3: out_strip(16x128) = attn_strip(16x2048) @ v(2048x128) ----
    // 8 waves; wave w owns output columns [w*16, w*16+16). K loop in chunks of 64,
    // staging registers double-buffer the v tile to overlap L2 reads with WMMA.
    const int half = lane >> 4;      // 0: lanes 0-15, 1: lanes 16-31
    const int col  = lane & 15;

    // staging ownership: thread owns column sn, k-pairs [skh*16 .. skh*16+15]
    const int sn  = tid & 127;
    const int skh = tid >> 7;

    v2h stage[16];
    auto load_chunk = [&](int k0) {
        #pragma unroll
        for (int q = 0; q < 16; ++q) {
            const int kq = skh * 16 + q;
            const float lo = vbh[(size_t)(k0 + 2 * kq)     * DVc + sn];
            const float hi = vbh[(size_t)(k0 + 2 * kq + 1) * DVc + sn];
            v2h p; p[0] = (_Float16)lo; p[1] = (_Float16)hi;
            stage[q] = p;
        }
    };
    auto store_chunk = [&]() {
        #pragma unroll
        for (int q = 0; q < 16; ++q) {
            const int kq = skh * 16 + q;
            *(v2h*)&sVt[sn * VT_STR + 2 * kq] = stage[q];   // packed ds_store_b32
        }
    };

    v8f acc = {};
    load_chunk(0);

    for (int k0 = 0; k0 < Lc; k0 += KC) {
        store_chunk();
        __syncthreads();
        if (k0 + KC < Lc) load_chunk(k0 + KC);   // overlap next L2 reads with WMMA

        #pragma unroll
        for (int s = 0; s < 2; ++s) {
            const int kb = k0 + s * 32;

            // A fragment (16x32 f16): lane (half,col) -> row M=col,
            // halfs 0..7 = K[kb+8*half .. +7], halfs 8..15 = K[kb+16+8*half .. +7]
            const float* ar = &sS[col * SS_STR + kb + (half ? 8 : 0)];
            const float4 x0 = *(const float4*)(ar);
            const float4 x1 = *(const float4*)(ar + 4);
            const float4 x2 = *(const float4*)(ar + 16);
            const float4 x3 = *(const float4*)(ar + 20);
            v16h af;
            af[0]  = (_Float16)x0.x; af[1]  = (_Float16)x0.y;
            af[2]  = (_Float16)x0.z; af[3]  = (_Float16)x0.w;
            af[4]  = (_Float16)x1.x; af[5]  = (_Float16)x1.y;
            af[6]  = (_Float16)x1.z; af[7]  = (_Float16)x1.w;
            af[8]  = (_Float16)x2.x; af[9]  = (_Float16)x2.y;
            af[10] = (_Float16)x2.z; af[11] = (_Float16)x2.w;
            af[12] = (_Float16)x3.x; af[13] = (_Float16)x3.y;
            af[14] = (_Float16)x3.z; af[15] = (_Float16)x3.w;

            // B fragment (32x16 f16): lane n = wave*16+col,
            // halfs 0..7 = v[kb + 8*half + j][n], halfs 8..15 = v[kb + 16 + 8*half + j-8][n]  (pre-transposed f16 in sVt)
            const _Float16* bp = &sVt[(wave * 16 + col) * VT_STR + s * 32 + (half ? 8 : 0)];
            v16h bf;
            *((v8h*)&bf)     = *(const v8h*)(bp);
            *((v8h*)&bf + 1) = *(const v8h*)(bp + 16);

            acc = __builtin_amdgcn_wmma_f32_16x16x32_f16(
                      false, af, false, bf, (short)0, acc, false, false);
        }
        __syncthreads();   // protect sVt before next staging commit
    }

    // Store C/D: VGPR g -> M = g (lanes 0-15) or g+8 (lanes 16-31), N = col
    #pragma unroll
    for (int g = 0; g < 8; ++g) {
        const int mrow = (lane < 16) ? g : (g + 8);
        out[((size_t)bh * Lc + i0 + mrow) * DVc + wave * 16 + col] = acc[g];
    }
}

extern "C" void kernel_launch(void* const* d_in, const int* in_sizes, int n_in,
                              void* d_out, int out_size, void* d_ws, size_t ws_size,
                              hipStream_t stream) {
    const float* vmat = (const float*)d_in[0];
    const float* a1   = (const float*)d_in[1];
    const float* a2   = (const float*)d_in[2];
    const int*   mask = (const int*)d_in[3];
    // d_in[4] = len_q (always 2048 here)

    float* out  = (float*)d_out;
    float* attn = out + (size_t)Bc * Hc * Lc * DVc;

    hipFuncSetAttribute((const void*)fra_fused_kernel,
                        hipFuncAttributeMaxDynamicSharedMemorySize,
                        (int)SMEM_BYTES);

    dim3 grid(Bc * Hc * (Lc / ROWS));
    dim3 block(NTHR);
    fra_fused_kernel<<<grid, block, SMEM_BYTES, stream>>>(vmat, a1, a2, mask, out, attn);
}
